// EncoderLayer_816043786694
// MI455X (gfx1250) — compile-verified, matrix operand order checked by reading, not run on hardware
//
#include <hip/hip_runtime.h>
#include <cstdint>
#include <cstddef>

// ---------------------------------------------------------------------------
// Types for CDNA5 WMMA (wave32, 16x16x32 f16 -> f32 accumulate)
// ---------------------------------------------------------------------------
typedef _Float16 f16;
typedef __attribute__((ext_vector_type(16))) _Float16 v16h;
typedef __attribute__((ext_vector_type(8)))  _Float16 v8h;
typedef __attribute__((ext_vector_type(8)))  float    v8f;

__device__ __forceinline__ v8f wmma16(v16h a, v16h b, v8f c) {
  // D = A(16x32) * B(32x16) + C, f32 accumulate
  return __builtin_amdgcn_wmma_f32_16x16x32_f16(
      /*neg_a=*/false, a, /*neg_b=*/false, b,
      /*c_mod=*/(short)0, c, /*reuse_a=*/false, /*reuse_b=*/false);
}

// 16-bit A/B fragment element pattern (ISA 7.12.2):
//   lane l: row/col = l&15 ; element e: k = (e/8)*16 + ((l&16)?8:0) + (e%8)
// => per lane: two contiguous 8-halfword (16B) loads at k-offsets klo and 16+klo.
__device__ __forceinline__ v16h load_frag(const f16* p, int klo) {
  v8h lo = *(const v8h*)(p + klo);
  v8h hi = *(const v8h*)(p + 16 + klo);
  v16h r;
#pragma unroll
  for (int i = 0; i < 8; ++i) { r[i] = lo[i]; r[i + 8] = hi[i]; }
  return r;
}

__device__ __forceinline__ v8f zero_acc() {
  v8f z = {0.f, 0.f, 0.f, 0.f, 0.f, 0.f, 0.f, 0.f};
  return z;
}

__device__ __forceinline__ v16h ones_frag() {
  v16h r;
#pragma unroll
  for (int i = 0; i < 16; ++i) r[i] = (f16)1.f;
  return r;
}

// max-reduction across the 16-lane half-wave that holds one C/D-tile row.
// ds_swizzle group-of-32 xor mode: imm = (xor_mask<<10) | (and_mask=0x1f)
__device__ __forceinline__ float rmax16(float x) {
  x = fmaxf(x, __int_as_float(__builtin_amdgcn_ds_swizzle(
                   __float_as_int(x), (1 << 10) | 0x1f)));
  x = fmaxf(x, __int_as_float(__builtin_amdgcn_ds_swizzle(
                   __float_as_int(x), (2 << 10) | 0x1f)));
  x = fmaxf(x, __int_as_float(__builtin_amdgcn_ds_swizzle(
                   __float_as_int(x), (4 << 10) | 0x1f)));
  x = fmaxf(x, __int_as_float(__builtin_amdgcn_ds_swizzle(
                   __float_as_int(x), (8 << 10) | 0x1f)));
  return x;
}

// ---------------------------------------------------------------------------
// Weight pack: f32 [N][K][TAPS] -> f16 [TAPS][N][K]  (TAPS=1: plain convert)
// ---------------------------------------------------------------------------
__global__ void pack_w_kernel(const float* __restrict__ src, f16* __restrict__ dst,
                              int N, int K, int TAPS) {
  int total = N * K * TAPS;
  int nk = N * K;
  for (int i = blockIdx.x * blockDim.x + threadIdx.x; i < total;
       i += gridDim.x * blockDim.x) {
    int t = i / nk;
    int r = i - t * nk;                    // r = n*K + k
    dst[i] = (f16)src[(size_t)r * TAPS + t];
  }
}

__global__ void zfill_f16(f16* __restrict__ p, int n) {
  for (int i = blockIdx.x * blockDim.x + threadIdx.x; i < n;
       i += gridDim.x * blockDim.x)
    p[i] = (f16)0.f;
}

// ---------------------------------------------------------------------------
// LayerNorm (torch-style: unbiased std, eps added to std), f32 in -> f16 out
// ---------------------------------------------------------------------------
__global__ void ln_kernel(const float* __restrict__ x, const float* __restrict__ g,
                          const float* __restrict__ b, f16* __restrict__ y, int D) {
  const int row = blockIdx.x;
  const float* xr = x + (size_t)row * D;
  __shared__ float sh1[256];
  __shared__ float sh2[256];
  float s = 0.f, s2 = 0.f;
  for (int i = threadIdx.x; i < D; i += 256) {
    float v = xr[i];
    s += v; s2 += v * v;
  }
  sh1[threadIdx.x] = s; sh2[threadIdx.x] = s2;
  __syncthreads();
  for (int o = 128; o > 0; o >>= 1) {
    if ((int)threadIdx.x < o) { sh1[threadIdx.x] += sh1[threadIdx.x + o];
                                sh2[threadIdx.x] += sh2[threadIdx.x + o]; }
    __syncthreads();
  }
  float mean = sh1[0] / (float)D;
  float var  = (sh2[0] - (float)D * mean * mean) / (float)(D - 1); // ddof=1
  float inv  = 1.f / (sqrtf(fmaxf(var, 0.f)) + 1e-6f);
  for (int i = threadIdx.x; i < D; i += 256) {
    float v = (xr[i] - mean) * inv;
    y[(size_t)row * D + i] = (f16)(g[i] * v + b[i]);
  }
}

// ---------------------------------------------------------------------------
// Generic WMMA GEMM: out[m,n] = epi( scale * sum_t A[m + t - TAPS/2, :] . Bw[t][n][:] )
//  - A: f16 [M][K]; rows shifted per tap; per-batch (S) boundary rows read a
//    zero-filled row (zrow) so the k-loop has no per-step selects.
//  - Bw: f16 [TAPS][N][K] packed (covers y = x @ W.T and k=3 'same' conv1d)
//  - epi: (+bias[n]) -> (ReLU?) -> (+resid[m,n]?) -> store
//  - OUT: 0 = f32 [M][N], 1 = f16 [M][N], 2 = f16 transposed [B][N][S] (for V)
// One wave computes a 32x64 tile: 2x4 accum tiles, 8 v_wmma per 32-deep k-step.
// ---------------------------------------------------------------------------
template <bool RELU, int OUT>
__global__ __launch_bounds__(32, 1)
void gemm_wmma(const f16* __restrict__ A, const f16* __restrict__ Bw,
               const f16* __restrict__ zrow,
               const float* __restrict__ bias, const float* __restrict__ resid,
               void* __restrict__ outv,
               int N, int K, int TAPS, int S, float scale) {
  const int lane = threadIdx.x;
  const int rA   = lane & 15;
  const int klo  = (lane & 16) ? 8 : 0;
  const int half8 = (lane & 16) ? 8 : 0;
  const int m0 = blockIdx.y * 32;
  const int n0 = blockIdx.x * 64;

  v8f acc[2][4];
#pragma unroll
  for (int mi = 0; mi < 2; ++mi)
#pragma unroll
    for (int ni = 0; ni < 4; ++ni) acc[mi][ni] = zero_acc();

  for (int t = 0; t < TAPS; ++t) {
    const int shift = t - (TAPS >> 1);   // TAPS=1 -> 0 ; TAPS=3 -> -1,0,1
    const f16* arow[2];
#pragma unroll
    for (int mi = 0; mi < 2; ++mi) {
      int gm = m0 + mi * 16 + rA;
      int spos = (gm % S) + shift;
      bool ok = ((unsigned)spos < (unsigned)S);
      arow[mi] = ok ? (A + (size_t)(gm + shift) * K) : zrow;
    }
    const f16* brow[4];
#pragma unroll
    for (int ni = 0; ni < 4; ++ni)
      brow[ni] = Bw + (size_t)t * N * K + (size_t)(n0 + ni * 16 + rA) * K;

    for (int k0 = 0; k0 < K; k0 += 32) {
      v16h af[2];
#pragma unroll
      for (int mi = 0; mi < 2; ++mi) af[mi] = load_frag(arow[mi] + k0, klo);
      v16h bfr[4];
#pragma unroll
      for (int ni = 0; ni < 4; ++ni) bfr[ni] = load_frag(brow[ni] + k0, klo);
#pragma unroll
      for (int mi = 0; mi < 2; ++mi)
#pragma unroll
        for (int ni = 0; ni < 4; ++ni)
          acc[mi][ni] = wmma16(af[mi], bfr[ni], acc[mi][ni]);
    }
  }

  // epilogue: C/D layout -> row = v + 8*(lane/16), col = lane&15
#pragma unroll
  for (int mi = 0; mi < 2; ++mi) {
#pragma unroll
    for (int ni = 0; ni < 4; ++ni) {
      const int col = n0 + ni * 16 + rA;
      const float bc = bias ? bias[col] : 0.f;
#pragma unroll
      for (int vv = 0; vv < 8; ++vv) {
        const int row = m0 + mi * 16 + vv + half8;
        float val = acc[mi][ni][vv] * scale + bc;
        if (RELU) val = fmaxf(val, 0.f);
        if (resid) val += resid[(size_t)row * N + col];
        if (OUT == 0) {
          ((float*)outv)[(size_t)row * N + col] = val;
        } else if (OUT == 1) {
          ((f16*)outv)[(size_t)row * N + col] = (f16)val;
        } else {  // f16, transposed per batch: [B][N][S]
          int bidx = row / S, srow = row % S;
          ((f16*)outv)[((size_t)bidx * N + col) * S + srow] = (f16)val;
        }
      }
    }
  }
}

// ---------------------------------------------------------------------------
// Flash attention: one wave per (b, h, 16-query tile). q pre-scaled by DH^-1/2.
// Scores: 16x32-key tiles via 4 WMMAs (K=DH=64); online softmax in f32 with
// row-max via ds_swizzle butterflies and row-sum via one P x ones WMMA;
// P transposed to A-fragment layout through LDS; P.V via 4 WMMAs against
// pre-transposed V ([B][D][S]) so V fragments are contiguous 16B loads.
// ---------------------------------------------------------------------------
__global__ __launch_bounds__(32, 1)
void attn_kernel(const f16* __restrict__ q, const f16* __restrict__ k,
                 const f16* __restrict__ vt,   // [B][D][S]
                 const unsigned char* __restrict__ mask,
                 f16* __restrict__ ctx, int S, int D, int DH) {
  const int lane = threadIdx.x;
  const int rA   = lane & 15;
  const int klo  = (lane & 16) ? 8 : 0;
  const int half8 = (lane & 16) ? 8 : 0;
  const int b = blockIdx.z, h = blockIdx.y;
  const int q0 = blockIdx.x * 16;
  const size_t bs = (size_t)b * S;
  const int hoff = h * DH;

  v16h qa[2];
#pragma unroll
  for (int kc = 0; kc < 2; ++kc)
    qa[kc] = load_frag(q + ((bs + q0 + rA) * D + hoff + kc * 32), klo);

  const v16h onef = ones_frag();

  float m_r[8], l_r[8];
  v8f oacc[4];
#pragma unroll
  for (int i = 0; i < 8; ++i) { m_r[i] = -1e30f; l_r[i] = 0.f; }
#pragma unroll
  for (int ni = 0; ni < 4; ++ni) oacc[ni] = zero_acc();

  __shared__ alignas(16) f16 pbuf[16 * 32];

  for (int kt = 0; kt < S; kt += 32) {
    v8f s0 = zero_acc(), s1 = zero_acc();
#pragma unroll
    for (int kc = 0; kc < 2; ++kc) {
      v16h kb0 = load_frag(k + ((bs + kt + rA) * D + hoff + kc * 32), klo);
      v16h kb1 = load_frag(k + ((bs + kt + 16 + rA) * D + hoff + kc * 32), klo);
      s0 = wmma16(qa[kc], kb0, s0);
      s1 = wmma16(qa[kc], kb1, s1);
    }
    const float mk0 = mask[bs + kt + rA]      ? -1e18f : 0.f;
    const float mk1 = mask[bs + kt + 16 + rA] ? -1e18f : 0.f;

#pragma unroll
    for (int vv = 0; vv < 8; ++vv) {
      float a0 = s0[vv] + mk0;
      float a1 = s1[vv] + mk1;
      float rowmax = rmax16(fmaxf(a0, a1));
      float mnew = fmaxf(m_r[vv], rowmax);
      float corr = __expf(m_r[vv] - mnew);
      m_r[vv] = mnew;
      l_r[vv] *= corr;                       // sum added back via ones-WMMA
      float p0 = __expf(a0 - mnew);
      float p1 = __expf(a1 - mnew);
      const int prow = vv + half8;
      pbuf[prow * 32 + rA]      = (f16)p0;
      pbuf[prow * 32 + 16 + rA] = (f16)p1;
#pragma unroll
      for (int ni = 0; ni < 4; ++ni) oacc[ni][vv] *= corr;
    }
    __syncthreads();
    v16h pa = load_frag(pbuf + rA * 32, klo);  // P as A-frag (16 rows x 32 keys)
    __syncthreads();

    // row sums of (f16-rounded) P: one WMMA; every column of D holds the sum
    v8f rs = wmma16(pa, onef, zero_acc());
#pragma unroll
    for (int vv = 0; vv < 8; ++vv) l_r[vv] += rs[vv];

#pragma unroll
    for (int ni = 0; ni < 4; ++ni) {
      v16h vf = load_frag(vt + ((size_t)(b * D + hoff + ni * 16 + rA)) * S + kt, klo);
      oacc[ni] = wmma16(pa, vf, oacc[ni]);
    }
  }

#pragma unroll
  for (int ni = 0; ni < 4; ++ni) {
#pragma unroll
    for (int vv = 0; vv < 8; ++vv) {
      const int row = vv + half8;
      float val = oacc[ni][vv] / l_r[vv];
      ctx[(bs + q0 + row) * D + hoff + ni * 16 + rA] = (f16)val;
    }
  }
}

// ---------------------------------------------------------------------------
// Host orchestration
// ---------------------------------------------------------------------------
extern "C" void kernel_launch(void* const* d_in, const int* in_sizes, int n_in,
                              void* d_out, int out_size, void* d_ws, size_t ws_size,
                              hipStream_t stream) {
  (void)in_sizes; (void)n_in; (void)out_size; (void)ws_size;
  constexpr int B = 4, S = 1024, D = 1024, H = 16, F = 4096;
  constexpr int M = B * S;        // 4096 token rows
  constexpr int DH = D / H;       // 64

  const float* x       = (const float*)d_in[0];
  const unsigned char* mask = (const unsigned char*)d_in[1];
  const float* gamma1  = (const float*)d_in[2];
  const float* beta1   = (const float*)d_in[3];
  const float* gamma2  = (const float*)d_in[4];
  const float* beta2   = (const float*)d_in[5];
  const float* Wq      = (const float*)d_in[6];
  const float* Wk      = (const float*)d_in[7];
  const float* Wv      = (const float*)d_in[8];
  const float* Wo      = (const float*)d_in[9];
  const float* c1w     = (const float*)d_in[10];
  const float* c1b     = (const float*)d_in[11];
  const float* c2w     = (const float*)d_in[12];
  const float* c2b     = (const float*)d_in[13];
  float* out           = (float*)d_out;

  // workspace layout (~160 MB)
  char* ws = (char*)d_ws;
  size_t off = 0;
  auto take = [&](size_t bytes) -> char* {
    char* p = ws + off;
    off += (bytes + 255) & ~(size_t)255;
    return p;
  };
  f16* WqH  = (f16*)take((size_t)D * D * sizeof(f16));
  f16* WkH  = (f16*)take((size_t)D * D * sizeof(f16));
  f16* WvH  = (f16*)take((size_t)D * D * sizeof(f16));
  f16* WoH  = (f16*)take((size_t)D * D * sizeof(f16));
  f16* W1H  = (f16*)take((size_t)3 * F * D * sizeof(f16));
  f16* W2H  = (f16*)take((size_t)3 * D * F * sizeof(f16));
  f16* xn1  = (f16*)take((size_t)M * D * sizeof(f16));
  f16* qh   = (f16*)take((size_t)M * D * sizeof(f16));
  f16* kh   = (f16*)take((size_t)M * D * sizeof(f16));
  f16* vtb  = (f16*)take((size_t)M * D * sizeof(f16));   // V transposed [B][D][S]
  f16* ctxh = (f16*)take((size_t)M * D * sizeof(f16));
  float* xres = (float*)take((size_t)M * D * sizeof(float));
  f16* xn2  = (f16*)take((size_t)M * D * sizeof(f16));
  f16* hbuf = (f16*)take((size_t)M * F * sizeof(f16));
  f16* zrow = (f16*)take((size_t)F * sizeof(f16));       // zero row (len >= max K)

  // 0) zero row for conv boundary taps
  zfill_f16<<<16, 256, 0, stream>>>(zrow, F);

  // 1) pack weights -> f16 [TAPS][N][K]
  pack_w_kernel<<<2048, 256, 0, stream>>>(Wq,  WqH, D, D, 1);
  pack_w_kernel<<<2048, 256, 0, stream>>>(Wk,  WkH, D, D, 1);
  pack_w_kernel<<<2048, 256, 0, stream>>>(Wv,  WvH, D, D, 1);
  pack_w_kernel<<<2048, 256, 0, stream>>>(Wo,  WoH, D, D, 1);
  pack_w_kernel<<<4096, 256, 0, stream>>>(c1w, W1H, F, D, 3);
  pack_w_kernel<<<4096, 256, 0, stream>>>(c2w, W2H, D, F, 3);

  const dim3 blk(32);
  const dim3 gD(D / 64, M / 32);   // N=1024 GEMMs
  const dim3 gF(F / 64, M / 32);   // N=4096 GEMM (conv1)

  // 2) LN1
  ln_kernel<<<M, 256, 0, stream>>>(x, gamma1, beta1, xn1, D);

  // 3) Q/K/V projections (q pre-scaled by DH^-1/2; V stored transposed)
  gemm_wmma<false, 1><<<gD, blk, 0, stream>>>(xn1, WqH, zrow, nullptr, nullptr,
                                              qh,  D, D, 1, S, 0.125f);
  gemm_wmma<false, 1><<<gD, blk, 0, stream>>>(xn1, WkH, zrow, nullptr, nullptr,
                                              kh,  D, D, 1, S, 1.0f);
  gemm_wmma<false, 2><<<gD, blk, 0, stream>>>(xn1, WvH, zrow, nullptr, nullptr,
                                              vtb, D, D, 1, S, 1.0f);

  // 4) flash attention -> ctx (written in [B,S,H*DH] layout)
  attn_kernel<<<dim3(S / 16, H, B), blk, 0, stream>>>(qh, kh, vtb, mask, ctxh,
                                                      S, D, DH);

  // 5) output projection + residual: xres = x + ctx @ Wo^T   (f32)
  gemm_wmma<false, 0><<<gD, blk, 0, stream>>>(ctxh, WoH, zrow, nullptr, x,
                                              xres, D, D, 1, S, 1.0f);

  // 6) LN2
  ln_kernel<<<M, 256, 0, stream>>>(xres, gamma2, beta2, xn2, D);

  // 7) conv1 (k=3, pad 1) + ReLU -> hbuf (f16)
  gemm_wmma<true, 1><<<gF, blk, 0, stream>>>(xn2, W1H, zrow, c1b, nullptr,
                                             hbuf, F, D, 3, S, 1.0f);

  // 8) conv2 (k=3, pad 1) + ReLU + residual -> out (f32)
  gemm_wmma<true, 0><<<gD, blk, 0, stream>>>(hbuf, W2H, zrow, c2b, xres,
                                             out, D, F, 3, S, 1.0f);
}
